// LSTMOptimizer_12678743457804
// MI455X (gfx1250) — matrix operand order repaired and checked by reading, not compiled, not run on hardware
//
#include <hip/hip_runtime.h>
#include <math.h>

typedef __attribute__((ext_vector_type(16))) _Float16 v16h;
typedef __attribute__((ext_vector_type(8)))  float    v8f;

#define HDIM  20
#define GATES 80

#define LOG2E  1.4426950408889634f
#define LN2_10 0.0693147180559945f   // ln(2)/10

// sigmoid(x) = 1/(1+e^-x) = v_rcp(1 + v_exp2(-x*log2e))
__device__ __forceinline__ float fast_sigmoid(float x) {
    return __builtin_amdgcn_rcpf(1.0f + __builtin_amdgcn_exp2f(-LOG2E * x));
}
// tanh(x) = 1 - 2/(1+e^{2x}) = 1 - 2*v_rcp(1 + v_exp2(2x*log2e))
__device__ __forceinline__ float fast_tanh(float x) {
    return 1.0f - 2.0f * __builtin_amdgcn_rcpf(1.0f + __builtin_amdgcn_exp2f(2.0f * LOG2E * x));
}

// preprocess: z1 = ax>=eps ? log(ax)/10 : -1 ; z2 = ax>=eps ? sign(x) : x*e^10
__device__ __forceinline__ void preprocess2(float x, float& z1, float& z2) {
    const float EPS = 4.5399929762484854e-05f; // e^-10
    const float BIG = 22026.465794806718f;     // e^+10
    float ax = fabsf(x);
    bool  c  = ax >= EPS;
    float lg = __builtin_amdgcn_logf(ax) * LN2_10;   // log2(ax)*ln2/10 = ln(ax)/10
    float sg = (x > 0.0f) ? 1.0f : ((x < 0.0f) ? -1.0f : 0.0f);
    z1 = c ? lg : -1.0f;
    z2 = c ? sg : x * BIG;
}

__global__ __launch_bounds__(256) void lstm_optimizer_kernel(
    const float* __restrict__ grad,  const float* __restrict__ loss,
    const float* __restrict__ f_tm1, const float* __restrict__ i_tm1,
    const float* __restrict__ theta_tm1, const float* __restrict__ delta_tm1,
    const float* __restrict__ h0,    const float* __restrict__ c0,
    const float* __restrict__ W_ih,  const float* __restrict__ W_hh,
    const float* __restrict__ b_ih,  const float* __restrict__ b_hh,
    const float* __restrict__ W_th_w, const float* __restrict__ W_th_b,
    const float* __restrict__ W_gr_w, const float* __restrict__ W_gr_b,
    float* __restrict__ out, int nElems)
{
    __shared__ float sGate[8][16 * GATES]; // per-wave 16x80 f32 gate scratch

    const int tid   = threadIdx.x;
    const int lane  = tid & 31;
    const int wvblk = tid >> 5;
    const int half  = lane >> 4;   // 0: K 0..7,16..23 ; 1: K 8..15,24..31
    const int l16   = lane & 15;
    float* gbuf = &sGate[wvblk][0];

    // ---- Build 5 B fragments (Wc is 24x80 zero-padded to K=32); wave-invariant ----
    // B layout (16-bit 32x16): lane -> col n%16; K = e + 8*half for e<8, K = 16 + 8*half + (e-8) for e>=8
    v16h  bfrag[5];
    float biasv[5];                // per-lane column bias, folded into WMMA C
    #pragma unroll
    for (int t = 0; t < 5; ++t) {
        const int g = t * 16 + l16;        // gate/output column 0..79
        biasv[t] = b_ih[g] + b_hh[g];
        #pragma unroll
        for (int e = 0; e < 16; ++e) {
            int K = (e < 8) ? (e + 8 * half) : (16 + 8 * half + (e - 8));
            float w;
            if (K < 4)       w = W_ih[g * 4 + K];          // x features
            else if (K < 24) w = W_hh[g * HDIM + (K - 4)]; // h0 features
            else             w = 0.0f;                     // K pad
            bfrag[t][e] = (_Float16)w;
        }
    }

    // ---- Hoist head weights into registers (per-lane slice [10*half, 10*half+10)) ----
    const int hbase = half * 10;
    float wth[10], wgr[10];
    #pragma unroll
    for (int k = 0; k < 10; ++k) {
        wth[k] = W_th_w[hbase + k];
        wgr[k] = W_gr_w[hbase + k];
    }
    const float wth20 = W_th_w[20], wth21 = W_th_w[21];
    const float wgr20 = W_gr_w[20], wgr21 = W_gr_w[21];
    const float bTh = W_th_b[0],    bGr = W_gr_b[0];

    const int nTiles = nElems >> 4; // N is a multiple of 16 (N = 2^22)
    const int gwave  = blockIdx.x * (blockDim.x >> 5) + wvblk;
    const int nWaves = gridDim.x * (blockDim.x >> 5);

    for (int tile = gwave; tile < nTiles; tile += nWaves) {
        const int row = tile * 16 + l16;          // element row for this lane pair
        const float* h0r = h0 + (long)row * HDIM;

        // features K0..3 from preprocess(grad), preprocess(loss)
        float g0 = grad[row];
        float f0, f1, f2, f3;
        preprocess2(g0, f0, f1);
        preprocess2(loss[row], f2, f3);

        // ---- A fragment: 16-bit 16x32 layout ----
        v16h a;
        if (half == 0) {
            a[0] = (_Float16)f0; a[1] = (_Float16)f1;
            a[2] = (_Float16)f2; a[3] = (_Float16)f3;   // K0..3
            #pragma unroll
            for (int e = 4; e < 8; ++e)  a[e] = (_Float16)h0r[e - 4];  // K4..7   -> h0[0..3]
            #pragma unroll
            for (int e = 8; e < 16; ++e) a[e] = (_Float16)h0r[e + 4];  // K16..23 -> h0[12..19]
        } else {
            #pragma unroll
            for (int e = 0; e < 8; ++e)  a[e] = (_Float16)h0r[e + 4];  // K8..15  -> h0[4..11]
            #pragma unroll
            for (int e = 8; e < 16; ++e) a[e] = (_Float16)0.0f;        // K24..31 pad
        }

        // ---- 5 WMMAs (bias pre-loaded in C): gates[16 rows][80] -> LDS ----
        #pragma unroll
        for (int t = 0; t < 5; ++t) {
            const float b = biasv[t];
            v8f d = {b, b, b, b, b, b, b, b};
            d = __builtin_amdgcn_wmma_f32_16x16x32_f16(
                    false, a, false, bfrag[t], (short)0, d, false, false);
            const int col = t * 16 + l16;         // D: lane -> column
            #pragma unroll
            for (int r = 0; r < 8; ++r) {
                const int M = r + 8 * half;       // D: vgpr r -> row r (+8 upper half)
                gbuf[M * GATES + col] = d[r];
            }
        }
        __builtin_amdgcn_wave_barrier();          // wave-local LDS: HW in-order, block reordering

        // ---- elementwise: lane handles row l16, h-range [10*half, 10*half+10) ----
        const float* gRow = gbuf + l16 * GATES;
        const float* c0r  = c0 + (long)(tile * 16 + l16) * HDIM;
        float accTh = 0.0f, accGr = 0.0f;
        #pragma unroll
        for (int k = 0; k < 10; ++k) {
            const int h = hbase + k;
            float ig = fast_sigmoid(gRow[h]);
            float fg = fast_sigmoid(gRow[20 + h]);
            float gg = fast_tanh   (gRow[40 + h]);
            float og = fast_sigmoid(gRow[60 + h]);
            float ct = fg * c0r[h] + ig * gg;
            float ht = og * fast_tanh(ct);
            accTh += ht * wth[k];
            accGr += ht * wgr[k];
        }
        // combine the two half-wave partial sums
        accTh += __shfl_xor(accTh, 16, 32);
        accGr += __shfl_xor(accGr, 16, 32);

        if (half == 0) {
            float th  = theta_tm1[row];
            float ft1 = f_tm1[row];
            float it1 = i_tm1[row];
            float dt1 = delta_tm1[row];
            float f_t = fast_sigmoid(accTh + th * wth20 + ft1 * wth21 + bTh);
            float i_t = fast_sigmoid(accGr + th * wgr20 + it1 * wgr21 + bGr);
            float delta_t = 0.1f * dt1 - i_t * g0;
            out[row] = f_t * th + delta_t;
        }
        __builtin_amdgcn_wave_barrier();          // keep next iter's stores after these loads
    }
}

extern "C" void kernel_launch(void* const* d_in, const int* in_sizes, int n_in,
                              void* d_out, int out_size, void* d_ws, size_t ws_size,
                              hipStream_t stream) {
    const float* grad      = (const float*)d_in[0];
    const float* loss      = (const float*)d_in[1];
    const float* f_tm1     = (const float*)d_in[2];
    const float* i_tm1     = (const float*)d_in[3];
    const float* theta_tm1 = (const float*)d_in[4];
    const float* delta_tm1 = (const float*)d_in[5];
    const float* h0        = (const float*)d_in[6];
    const float* c0        = (const float*)d_in[7];
    const float* W_ih      = (const float*)d_in[8];
    const float* W_hh      = (const float*)d_in[9];
    const float* b_ih      = (const float*)d_in[10];
    const float* b_hh      = (const float*)d_in[11];
    const float* W_th_w    = (const float*)d_in[12];
    const float* W_th_b    = (const float*)d_in[13];
    const float* W_gr_w    = (const float*)d_in[14];
    const float* W_gr_b    = (const float*)d_in[15];
    float* out = (float*)d_out;

    const int n = in_sizes[0];          // N elements (grad_t flat count)
    const int blocks = 4096;            // grid-stride: 32768 waves, ~8 tiles each
    lstm_optimizer_kernel<<<blocks, 256, 0, stream>>>(
        grad, loss, f_tm1, i_tm1, theta_tm1, delta_tm1, h0, c0,
        W_ih, W_hh, b_ih, b_hh, W_th_w, W_th_b, W_gr_w, W_gr_b,
        out, n);
}
